// FlatSquare7x7NNUEv3_47519518163399
// MI455X (gfx1250) — matrix operand order repaired and checked by reading, not compiled, not run on hardware
//
#include <hip/hip_runtime.h>
#include <hip/hip_bf16.h>

// ---------------------------------------------------------------------------
// FlatSquare7x7 NNUE forward for MI455X (gfx1250, wave32, WMMA).
// Compute-bound (~63 GFLOP vs ~5MB I/O) -> map 128-ch convs to
// v_wmma_f32_16x16x32_f16. Whole 7-layer mapping fused per workgroup in LDS.
// ---------------------------------------------------------------------------

typedef __attribute__((ext_vector_type(16))) _Float16 v16h;
typedef __attribute__((ext_vector_type(8)))  float    v8f;

struct alignas(16) I4 { int x, y, z, w; };   // POD 16B chunk (union-safe)

#define BATCH   8192
#define TB      16          // boards per workgroup
#define STRIDE  136         // LDS row stride in halves (128 + 8 pad)

// PATTERNS: y0,y1,x0,x1,rot_k,group
__constant__ int PATY0[12] = {0,0,4,3, 0,0,3,4, 0,2,3,2};
__constant__ int PATY1[12] = {3,4,7,7, 4,3,7,7, 4,5,7,5};
__constant__ int PATX0[12] = {0,4,3,0, 0,3,4,0, 2,3,2,0};
__constant__ int PATX1[12] = {4,7,7,3, 3,7,7,4, 5,7,5,4};
__constant__ int PATK [12] = {0,1,2,3, 3,0,1,2, 3,0,1,2};
__constant__ int PATG [12] = {0,0,0,0, 1,1,1,1, 2,2,2,2};

// Geometry of layers 1..6 (0-indexed here as 0..5): K_total, out_ch, kh, kw
__constant__ int L_K [6] = {128,512,128,256,128,128};
__constant__ int L_CO[6] = {128,128,128,128,128, 32};
__constant__ int L_KH[6] = {1,2,1,1,1,1};
__constant__ int L_KW[6] = {1,2,1,2,1,1};

__device__ __forceinline__ float mishf(float x) {
    // mish(x) = x * tanh(softplus(x)) = x * ((1+e^x)^2 - 1)/((1+e^x)^2 + 1)
    float t = __expf(fminf(x, 30.0f));
    float u = 1.0f + t; u = u * u;
    return x * (u - 1.0f) / (u + 1.0f);
}

// ---------------------------------------------------------------------------
// Weight prep: fold BN, emit f16 weights in WMMA B-fragment order.
// B (32x16, KxN) own contiguous k order (as A): lane L holds column n=L&15, K half = (L&16?16:0)+h.
// Fragment storage: frag(ntile,kstep) 512 halves, per-lane 16 contiguous halves.
// ---------------------------------------------------------------------------
struct PrepArgs {
    const float* W [3][6]; const float* b [3][6];
    const float* ga[3][6]; const float* be[3][6];
    const float* mu[3][6]; const float* var[3][6];
    _Float16* frags[3][6]; float* bias[3][6];
};

__global__ void prep_weights(PrepArgs pa) {
    int blk = blockIdx.x;            // 0..17
    int mp  = blk / 6, l = blk % 6;
    int K = L_K[l], CO = L_CO[l], kh = L_KH[l], kw = L_KW[l];
    __shared__ float s_s[128];
    int tid = threadIdx.x;
    if (tid < CO) {
        float s = pa.ga[mp][l][tid] * rsqrtf(pa.var[mp][l][tid] + 1e-5f);
        s_s[tid] = s;
        pa.bias[mp][l][tid] = s * (pa.b[mp][l][tid] - pa.mu[mp][l][tid]) + pa.be[mp][l][tid];
    }
    __syncthreads();
    int nt = CO >> 4, kt = K >> 5;
    int total = nt * kt * 512;
    const float* W = pa.W[mp][l];
    _Float16* fr = pa.frags[mp][l];
    for (int e = tid; e < total; e += 256) {
        int frag = e >> 9, within = e & 511;
        int lane = within >> 4, h = within & 15;
        int ntile = frag / kt, kstep = frag % kt;
        int n  = lane & 15;
        int k  = kstep * 32 + ((lane & 16) ? 16 : 0) + h;
        int co = ntile * 16 + n;
        int tap = k >> 7, ci = k & 127;      // im2col: k = tap*128 + ci
        int ty = tap / kw, tx = tap % kw;
        float w = W[((co * 128 + ci) * kh + ty) * kw + tx] * s_s[co];
        fr[e] = (_Float16)w;
    }
}

// ---------------------------------------------------------------------------
// Generic fused WMMA GEMM layer (LDS activations -> LDS / global feature out).
// MODE: 0 = identity rows (1x1 conv), 1 = 2x2 conv on 2x3 grid, 2 = 1x2 conv.
// ACT : 1 = mish -> f16 LDS, 2 = clip + round-quant(1/128) -> f32 global.
// ---------------------------------------------------------------------------
template<int MODE, int ACT, int KSTEPS>
__device__ void gemm_layer(const _Float16* __restrict__ frags,
                           const float* __restrict__ bias,
                           const _Float16* src, _Float16* dst,
                           float* __restrict__ featOut,
                           int rowtiles, int coltiles)
{
    int wave = threadIdx.x >> 5;
    int lane = threadIdx.x & 31;
    int n    = lane & 15;
    int kb   = (lane & 16) ? 16 : 0;  // A-fragment K base (contiguous k = 16*half + s, as B)
    int mhi  = (lane & 16) ? 8 : 0;   // C/D row base
    int tasks = rowtiles * coltiles;
    for (int t = wave; t < tasks; t += 8) {
        int rowtile = t / coltiles;
        int ntile   = t % coltiles;
        v8f acc = {0.f,0.f,0.f,0.f,0.f,0.f,0.f,0.f};
        const _Float16* bbase = frags + (size_t)ntile * KSTEPS * 512 + lane * 16;
        int m = rowtile * 16 + n;                   // A row index (M = lane&15)
#pragma unroll
        for (int kk = 0; kk < KSTEPS; ++kk) {
            union { I4 q[2]; v16h h; } bu;
            bu.q[0] = *(const I4*)(bbase + kk * 512);
            bu.q[1] = *(const I4*)(bbase + kk * 512 + 8);
            union { I4 q[2]; v16h h; } au;
#pragma unroll
            for (int seg = 0; seg < 2; ++seg) {
                int k0  = kk * 32 + seg * 8 + kb;   // first of 8 contiguous K
                int tap = k0 >> 7;
                int ch  = k0 & 127;
                int r;
                if (MODE == 0)      r = m;
                else if (MODE == 1) {               // 2x2 conv over 2x3 spatial
                    int bd = m >> 1, op = m & 1;
                    r = bd * 6 + (tap >> 1) * 3 + op + (tap & 1);
                } else {                            // 1x2 conv over 1x2 spatial
                    r = m * 2 + tap;
                }
                au.q[seg] = *(const I4*)(src + r * STRIDE + ch);
            }
            acc = __builtin_amdgcn_wmma_f32_16x16x32_f16(
                      false, au.h, false, bu.h, (short)0, acc, false, false);
        }
        int co = ntile * 16 + n;
        float bv = bias[co];
        int mbase = rowtile * 16 + mhi;
#pragma unroll
        for (int r = 0; r < 8; ++r) {
            float v = acc[r] + bv;
            int mm = mbase + r;
            if (ACT == 1) {
                dst[mm * STRIDE + co] = (_Float16)mishf(v);
            } else if (ACT == 2) {
                v = fminf(fmaxf(v, -1.0f), 127.0f / 128.0f);
                float q = rintf(v * 128.0f);
                q = fminf(fmaxf(q, -128.0f), 127.0f);
                featOut[mm * 32 + co] = q * (1.0f / 128.0f);
            } else {
                dst[mm * STRIDE + co] = (_Float16)v;
            }
        }
    }
}

// ---------------------------------------------------------------------------
// Fused mapping kernel: one WG = (pattern, 16-board tile). All 7 layers in LDS.
// ---------------------------------------------------------------------------
struct MapArgs {
    const float* board;
    const float* W1[3]; const float* b1[3];        // layer 0 (raw, no BN)
    const _Float16* frags[3][6];
    const float* bias[3][6];
    float* featAll;                                 // [12][BATCH][32] f32
};

__global__ void __launch_bounds__(256) map_kernel(MapArgs a) {
    int pat = blockIdx.x / (BATCH / TB);
    int b0  = (blockIdx.x % (BATCH / TB)) * TB;
    int g   = PATG[pat];
    int y0 = PATY0[pat], x0 = PATX0[pat];
    int hs = PATY1[pat] - y0, ws = PATX1[pat] - x0;
    int rk = PATK[pat];

    __shared__ float    chunk[TB][2][3][4];
    __shared__ _Float16 actA[96 * STRIDE];
    __shared__ _Float16 actB[96 * STRIDE];

    int tid = threadIdx.x;
    // Stage rotated chunks: rot90(board[:, :, y0:y1, x0:x1], rk) -> [2,3,4]
    for (int idx = tid; idx < TB * 24; idx += 256) {
        int b = idx / 24, rem = idx % 24;
        int c = rem / 12, pos = rem % 12;
        int i = pos >> 2, j = pos & 3;
        int si, sj;
        if      (rk == 0) { si = i;          sj = j;          }
        else if (rk == 1) { si = j;          sj = ws - 1 - i; }
        else if (rk == 2) { si = hs - 1 - i; sj = ws - 1 - j; }
        else              { si = hs - 1 - j; sj = i;          }
        chunk[b][c][i][j] =
            a.board[((size_t)(b0 + b) * 2 + c) * 49 + (y0 + si) * 7 + (x0 + sj)];
    }
    __syncthreads();

    // Layer 0: conv 2->128, 2x2 on 3x4 -> 2x3, mish. K=8: VALU, not WMMA.
    const float* W1 = a.W1[g];
    const float* b1 = a.b1[g];
    for (int idx = tid; idx < TB * 6 * 128; idx += 256) {
        int b = idx / (6 * 128); int rem = idx % (6 * 128);
        int pos = rem >> 7; int co = rem & 127;
        int oy = pos / 3, ox = pos % 3;
        float acc = b1[co];
#pragma unroll
        for (int c = 0; c < 2; c++)
#pragma unroll
            for (int dy = 0; dy < 2; dy++)
#pragma unroll
                for (int dx = 0; dx < 2; dx++)
                    acc += W1[co * 8 + c * 4 + dy * 2 + dx] * chunk[b][c][oy + dy][ox + dx];
        actA[(b * 6 + pos) * STRIDE + co] = (_Float16)mishf(acc);
    }
    __syncthreads();

    // Layer 1: 1x1 128->128, rows = 96 (6 tiles)
    gemm_layer<0, 1, 4 >(a.frags[g][0], a.bias[g][0], actA, actB, nullptr, 6, 8);
    __syncthreads();
    // Layer 2: 2x2 conv 128->128, rows = 32, K = 512 (im2col at A-load)
    gemm_layer<1, 1, 16>(a.frags[g][1], a.bias[g][1], actB, actA, nullptr, 2, 8);
    __syncthreads();
    // Layer 3: 1x1, rows = 32
    gemm_layer<0, 1, 4 >(a.frags[g][2], a.bias[g][2], actA, actB, nullptr, 2, 8);
    __syncthreads();
    // Layer 4: 1x2 conv, rows = 16, K = 256
    gemm_layer<2, 1, 8 >(a.frags[g][3], a.bias[g][3], actB, actA, nullptr, 1, 8);
    __syncthreads();
    // Layer 5: 1x1, rows = 16
    gemm_layer<0, 1, 4 >(a.frags[g][4], a.bias[g][4], actA, actB, nullptr, 1, 8);
    __syncthreads();
    // Layer 6: 1x1 128->32, no activation; clip + fake_quant(128) -> feat
    float* featPat = a.featAll + ((size_t)pat * BATCH + b0) * 32;
    gemm_layer<0, 2, 4 >(a.frags[g][5], a.bias[g][5], actB, nullptr, featPat, 1, 2);
}

// ---------------------------------------------------------------------------
// Value head: feature combine + 3 quantized linear layers. Tiny: VALU.
// ---------------------------------------------------------------------------
struct ValArgs {
    const float* featAll;
    const float* VW[3]; const float* Vb[3];
    float* out;
};

__global__ void __launch_bounds__(256) value_kernel(ValArgs a) {
    __shared__ float W1q[64 * 128];
    __shared__ float W2q[32 * 64];
    __shared__ float W3q[3 * 32];
    __shared__ float bq[64 + 32 + 3];
    int tid = threadIdx.x;
    const float s1 = 256.0f, s2 = 128.0f, s3 = 128.0f;
    for (int i = tid; i < 64 * 128; i += 256)
        W1q[i] = fminf(fmaxf(rintf(a.VW[0][i] * s1), -128.f), 127.f) / s1;
    for (int i = tid; i < 32 * 64; i += 256)
        W2q[i] = fminf(fmaxf(rintf(a.VW[1][i] * s2), -128.f), 127.f) / s2;
    for (int i = tid; i < 3 * 32; i += 256)
        W3q[i] = fminf(fmaxf(rintf(a.VW[2][i] * s3), -128.f), 127.f) / s3;
    if (tid < 64) bq[tid]      = rintf(a.Vb[0][tid] * s1 * 128.f) / (s1 * 128.f);
    if (tid < 32) bq[64 + tid] = rintf(a.Vb[1][tid] * s2 * 128.f) / (s2 * 128.f);
    if (tid < 3)  bq[96 + tid] = rintf(a.Vb[2][tid] * s3 * 128.f) / (s3 * 128.f);
    __syncthreads();

    int b = blockIdx.x * 256 + tid;
    float v1[64];
#pragma unroll
    for (int co = 0; co < 64; co++) v1[co] = bq[co];
    for (int k = 0; k < 128; k++) {
        int pidx = k >> 5, c = k & 31;
        size_t base = ((size_t)pidx * BATCH + b) * 32 + c;
        float f1 = a.featAll[base];
        float f2 = a.featAll[base + (size_t)4 * BATCH * 32];
        float fm = a.featAll[base + (size_t)8 * BATCH * 32];
        float s  = (f1 + f2 + 1.0f / 128.0f) * 0.5f;
        float q  = floorf(s * 128.0f);
        q = fminf(fmaxf(q, -128.0f), 127.0f);
        float feat = q * (1.0f / 128.0f) + fm;
        feat = fminf(fmaxf(feat, -1.0f), 127.0f / 128.0f);   // clip before layer 0
        for (int co = 0; co < 64; co++) v1[co] += W1q[co * 128 + k] * feat;
    }
    float v2[32];
#pragma unroll
    for (int co = 0; co < 32; co++) v2[co] = bq[64 + co];
    for (int k = 0; k < 64; k++) {
        float x = fminf(fmaxf(v1[k], 0.0f), 127.0f / 128.0f);
        for (int co = 0; co < 32; co++) v2[co] += W2q[co * 64 + k] * x;
    }
    float v3[3] = {bq[96], bq[97], bq[98]};
    for (int k = 0; k < 32; k++) {
        float x = fminf(fmaxf(v2[k], 0.0f), 127.0f / 128.0f);
        for (int co = 0; co < 3; co++) v3[co] += W3q[co * 32 + k] * x;
    }
    a.out[(size_t)b * 3 + 0] = v3[0];
    a.out[(size_t)b * 3 + 1] = v3[1];
    a.out[(size_t)b * 3 + 2] = v3[2];
}

__global__ void zero_kernel(float* p, int n) {
    int i = blockIdx.x * 256 + threadIdx.x;
    if (i < n) p[i] = 0.0f;
}

// ---------------------------------------------------------------------------
extern "C" void kernel_launch(void* const* d_in, const int* in_sizes, int n_in,
                              void* d_out, int out_size, void* d_ws, size_t ws_size,
                              hipStream_t stream) {
    (void)in_sizes; (void)n_in; (void)out_size; (void)ws_size;
    int idx = 0;
    const float* board = (const float*)d_in[idx++];
    MapArgs ma{}; PrepArgs pa{}; ValArgs va{};
    for (int m = 0; m < 3; m++) {
        for (int l = 0; l < 7; l++) {
            const float* W = (const float*)d_in[idx++];
            const float* b = (const float*)d_in[idx++];
            if (l == 0) { ma.W1[m] = W; ma.b1[m] = b; }
            else {
                pa.W [m][l-1] = W; pa.b[m][l-1] = b;
                pa.ga[m][l-1] = (const float*)d_in[idx++];
                pa.be[m][l-1] = (const float*)d_in[idx++];
                pa.mu[m][l-1] = (const float*)d_in[idx++];
                pa.var[m][l-1]= (const float*)d_in[idx++];
            }
        }
    }
    for (int i = 0; i < 3; i++) {
        va.VW[i] = (const float*)d_in[idx++];
        va.Vb[i] = (const float*)d_in[idx++];
    }

    // Workspace carve (256B aligned): featAll + prepped weights/biases
    char* ws = (char*)d_ws;
    size_t off = 0;
    auto carve = [&](size_t bytes) -> char* {
        char* p = ws + off;
        off = (off + bytes + 255) & ~(size_t)255;
        return p;
    };
    float* featAll = (float*)carve((size_t)12 * BATCH * 32 * sizeof(float));
    static const int KT_[6] = {128,512,128,256,128,128};
    static const int CO_[6] = {128,128,128,128,128, 32};
    for (int m = 0; m < 3; m++) {
        for (int l = 0; l < 6; l++) {
            size_t halves = (size_t)(CO_[l] / 16) * (KT_[l] / 32) * 512;
            _Float16* f = (_Float16*)carve(halves * 2);
            float* bb = (float*)carve((size_t)CO_[l] * 4);
            pa.frags[m][l] = f; pa.bias[m][l] = bb;
            ma.frags[m][l] = f; ma.bias[m][l] = bb;
        }
    }
    ma.board = board; ma.featAll = featAll;
    va.featAll = featAll; va.out = (float*)d_out;

    prep_weights<<<18, 256, 0, stream>>>(pa);
    map_kernel<<<12 * (BATCH / TB), 256, 0, stream>>>(ma);
    value_kernel<<<BATCH / 256, 256, 0, stream>>>(va);
    int npol = BATCH * 49;
    zero_kernel<<<(npol + 255) / 256, 256, 0, stream>>>((float*)d_out + (size_t)BATCH * 3, npol);
}
